// TransformerBlock_42228118454552
// MI455X (gfx1250) — matrix operand order repaired and checked by reading, not compiled, not run on hardware
//
#include <hip/hip_runtime.h>
#include <math.h>

#define EMB 1024
#define NHEAD 16
#define HD 64
#define FFDIM 4096
#define BATCH 2
#define SEQ 2048
#define MROWS (BATCH * SEQ) // 4096

typedef __attribute__((ext_vector_type(16))) __bf16 v16bf;
typedef __attribute__((ext_vector_type(8)))  __bf16 v8bf;
typedef __attribute__((ext_vector_type(8)))  float  v8f;

union U16 { v16bf v; v8bf h[2]; __bf16 e[16]; };

__device__ __forceinline__ v8f wmma_bf16(v16bf a, v16bf b, v8f c) {
  // 8 args: (neg_a, A, neg_b, B, c_mod, C, reuse_a, reuse_b)
  return __builtin_amdgcn_wmma_f32_16x16x32_bf16(false, a, false, b, (short)0, c,
                                                 false, false);
}

// CDNA5 async global->LDS copy (ASYNCcnt-tracked), 16 bytes per lane.
// dst operand is the per-lane LDS byte address; flat addr low 32 bits map to LDS.
__device__ __forceinline__ void async_ld_b128(unsigned lds_addr, const void* gaddr) {
  asm volatile("global_load_async_to_lds_b128 %0, %1, off"
               :: "v"(lds_addr), "v"(gaddr)
               : "memory");
}
__device__ __forceinline__ void wait_asynccnt0() {
  asm volatile("s_wait_asynccnt 0x0" ::: "memory");
}

// ---------------------------------------------------------------- fp32 -> bf16
__global__ void cvt_bf16_k(const float* __restrict__ s, __bf16* __restrict__ d, int n) {
  int i = blockIdx.x * blockDim.x + threadIdx.x;
  int stride = gridDim.x * blockDim.x;
  for (; i < n; i += stride) d[i] = (__bf16)s[i];
}

// ---------------------------------------------------------------- LayerNorm
__global__ __launch_bounds__(256)
void ln_bf16_k(const float* __restrict__ x, const float* __restrict__ g,
               const float* __restrict__ bta, __bf16* __restrict__ out) {
  __shared__ float r1[256];
  __shared__ float r2[256];
  int row = blockIdx.x, tid = threadIdx.x;
  const float* xr = x + (size_t)row * EMB;
  float v[4], s = 0.f, s2 = 0.f;
#pragma unroll
  for (int i = 0; i < 4; ++i) {
    v[i] = xr[tid + i * 256];
    s += v[i];
    s2 += v[i] * v[i];
  }
  r1[tid] = s; r2[tid] = s2;
  __syncthreads();
  for (int off = 128; off > 0; off >>= 1) {
    if (tid < off) { r1[tid] += r1[tid + off]; r2[tid] += r2[tid + off]; }
    __syncthreads();
  }
  float mu  = r1[0] * (1.0f / EMB);
  float var = r2[0] * (1.0f / EMB) - mu * mu;
  float rs  = rsqrtf(var + 1e-5f);
#pragma unroll
  for (int i = 0; i < 4; ++i) {
    int c = tid + i * 256;
    out[(size_t)row * EMB + c] = (__bf16)((v[i] - mu) * rs * g[c] + bta[c]);
  }
}

// ---------------------------------------------------------------- GEMM (bf16 WMMA)
// block tile 128x128, 8 waves as 4(M) x 2(N), wave tile 32x64 -> 8 wmma per K-step.
// Double-buffered LDS: tile i+1 streams in (async DMA for A, VMEM->LDS transpose
// for W) while tile i's WMMAs execute; one barrier + one asynccnt wait per step.
#define GM 128
#define GN 128
#define GK 32
enum { MODE_BF16 = 0, MODE_QKV = 1, MODE_GELU = 2, MODE_RES = 3 };

__global__ __launch_bounds__(256)
void gemm_wmma_k(const __bf16* __restrict__ A, const __bf16* __restrict__ W,
                 const float* __restrict__ bias, int M, int N, int K,
                 int mode, float scale, const float* __restrict__ alphap,
                 const float* __restrict__ res, float* __restrict__ outf,
                 __bf16* __restrict__ outb) {
  __shared__ __attribute__((aligned(64))) __bf16 Alds[2][GM * GK]; // 2 x 8 KB
  __shared__ __attribute__((aligned(64))) __bf16 Wt[2][GN * GK];   // 2 x 8 KB, [n][k]
  int tid = threadIdx.x;
  int m0 = blockIdx.y * GM;
  int n0 = blockIdx.x * GN;
  int wid = tid >> 5, lane = tid & 31;
  int wm = wid >> 1, wn = wid & 1;       // 4 x 2 waves -> 128 x 128 tile
  int l = lane & 15, hi = lane >> 4;

  int arow = tid >> 1, achunk = tid & 1; // A tile: 128 rows x 32 cols, 16 elems/thread
  int wkrow = tid >> 3, wnc = tid & 7;   // W tile: 32 rows x 128 cols, 16 elems/thread

  v8f acc[2][4] = {};

  unsigned aB[2] = {
      (unsigned)(uintptr_t)&Alds[0][arow * GK + achunk * 16],
      (unsigned)(uintptr_t)&Alds[1][arow * GK + achunk * 16]};
  const __bf16* agp = &A[(size_t)(m0 + arow) * K + achunk * 16];
  const __bf16* wgp = &W[(size_t)wkrow * N + n0 + wnc * 16];
  int nk = K >> 5; // K-steps

  // ---- prologue: stage tile 0 into buffer 0
  async_ld_b128(aB[0], agp);
  async_ld_b128(aB[0] + 16, agp + 8);
  {
    U16 w0;
    w0.v = *(const v16bf*)wgp;
#pragma unroll
    for (int i = 0; i < 16; ++i) Wt[0][(wnc * 16 + i) * GK + wkrow] = w0.e[i];
  }
  wait_asynccnt0();
  __syncthreads();

  for (int it = 0; it < nk; ++it) {
    int p = it & 1;
    bool more = (it + 1) < nk;
    U16 wreg;
    if (more) {
      // kick off tile it+1 while tile it computes
      const __bf16* an = agp + (size_t)(it + 1) * GK;
      async_ld_b128(aB[p ^ 1], an);
      async_ld_b128(aB[p ^ 1] + 16, an + 8);
      wreg.v = *(const v16bf*)(wgp + (size_t)(it + 1) * GK * N);
      if (it + 2 < nk)
        __builtin_prefetch(wgp + (size_t)(it + 2) * GK * N, 0, 1);
    }

    // ---- compute tile it from buffer p
    U16 af[2], bfm[4];
#pragma unroll
    for (int mt = 0; mt < 2; ++mt) {
      int r = wm * 32 + mt * 16 + l;
      af[mt].h[0] = *(const v8bf*)&Alds[p][r * GK + hi * 8];
      af[mt].h[1] = *(const v8bf*)&Alds[p][r * GK + hi * 8 + 16];
    }
#pragma unroll
    for (int nt = 0; nt < 4; ++nt) {
      int c = wn * 64 + nt * 16 + l;
      bfm[nt].h[0] = *(const v8bf*)&Wt[p][c * GK + hi * 8];
      bfm[nt].h[1] = *(const v8bf*)&Wt[p][c * GK + hi * 8 + 16];
    }
#pragma unroll
    for (int mt = 0; mt < 2; ++mt)
#pragma unroll
      for (int nt = 0; nt < 4; ++nt)
        acc[mt][nt] = wmma_bf16(af[mt].v, bfm[nt].v, acc[mt][nt]);

    if (more) {
      // W global load latency was hidden under the WMMAs; transpose into buf p^1
#pragma unroll
      for (int i = 0; i < 16; ++i) Wt[p ^ 1][(wnc * 16 + i) * GK + wkrow] = wreg.e[i];
    }
    wait_asynccnt0();
    __syncthreads();
  }

  float alpha = (mode == MODE_RES) ? alphap[0] : 0.f;
#pragma unroll
  for (int mt = 0; mt < 2; ++mt)
#pragma unroll
    for (int nt = 0; nt < 4; ++nt)
#pragma unroll
      for (int g = 0; g < 8; ++g) {
        int m = m0 + wm * 32 + mt * 16 + g + hi * 8;
        int n = n0 + wn * 64 + nt * 16 + l;
        float v = acc[mt][nt][g] + bias[n];
        size_t idx = (size_t)m * N + n;
        if (mode == MODE_QKV) {
          int b = m / SEQ, t = m % SEQ;
          int h = n >> 6, d = n & 63;
          outb[(((size_t)(b * NHEAD + h)) * SEQ + t) * HD + d] = (__bf16)(v * scale);
        } else if (mode == MODE_GELU) {
          float gv = 0.5f * v * (1.0f + erff(v * 0.70710678f));
          outb[idx] = (__bf16)gv;
        } else if (mode == MODE_RES) {
          outf[idx] = res[idx] + alpha * v;
        } else {
          outb[idx] = (__bf16)v;
        }
      }
}

// ---------------------------------------------------------------- flash attention
// one wave per (b, h, 16-query-row tile); streams 32-key blocks with online softmax
__global__ __launch_bounds__(32)
void attn_fwd_k(const __bf16* __restrict__ Q, const __bf16* __restrict__ Kk,
                const __bf16* __restrict__ V, __bf16* __restrict__ AO) {
  __shared__ __attribute__((aligned(64))) float  Sb[16 * 32];
  __shared__ __attribute__((aligned(64))) __bf16 Pb[16 * 32];
  __shared__ float rowv[16];

  int lane = threadIdx.x, l = lane & 15, hi = lane >> 4;
  int qt = blockIdx.x, bh = blockIdx.y;
  int b = bh / NHEAD, h = bh % NHEAD;
  size_t base = (size_t)bh * SEQ * HD;
  int q0 = qt * 16;

  // Q fragments for d = [0,32) and [32,64)  (q already scaled by 1/sqrt(64))
  U16 qf[2];
#pragma unroll
  for (int dd = 0; dd < 2; ++dd) {
    const __bf16* p = &Q[base + (size_t)(q0 + l) * HD + dd * 32 + hi * 16];
    qf[dd].h[0] = *(const v8bf*)p;
    qf[dd].h[1] = *(const v8bf*)(p + 8);
  }

  v8f o[4] = {};
  float m_r = -1e30f, l_r = 0.f;

  for (int kb = 0; kb < q0 + 16; kb += 32) {
    // S = Q * K^T  (16 x 32 scores)
    v8f s0 = {}, s1 = {};
#pragma unroll
    for (int dd = 0; dd < 2; ++dd) {
      U16 kf0, kf1;
      const __bf16* p0 = &Kk[base + (size_t)(kb + l) * HD + dd * 32 + hi * 16];
      kf0.h[0] = *(const v8bf*)p0;
      kf0.h[1] = *(const v8bf*)(p0 + 8);
      const __bf16* p1 = &Kk[base + (size_t)(kb + 16 + l) * HD + dd * 32 + hi * 16];
      kf1.h[0] = *(const v8bf*)p1;
      kf1.h[1] = *(const v8bf*)(p1 + 8);
      s0 = wmma_bf16(qf[dd].v, kf0.v, s0);
      s1 = wmma_bf16(qf[dd].v, kf1.v, s1);
    }
#pragma unroll
    for (int g = 0; g < 8; ++g) {
      int m = g + hi * 8;
      Sb[m * 32 + l]      = s0[g];
      Sb[m * 32 + 16 + l] = s1[g];
    }
    __syncthreads();

    // online softmax, lanes 0..15 own one row each (causal mask applied here)
    if (lane < 16) {
      int r = lane, qg = q0 + r;
      float sv[32];
      float mmax = m_r;
#pragma unroll
      for (int c = 0; c < 32; ++c) {
        float xv = (kb + c <= qg) ? Sb[r * 32 + c] : -1e30f;
        sv[c] = xv;
        mmax = fmaxf(mmax, xv);
      }
      float sc = __expf(m_r - mmax);
      float ls = l_r * sc;
#pragma unroll
      for (int c = 0; c < 32; ++c) {
        float p = (kb + c <= qg) ? __expf(sv[c] - mmax) : 0.f;
        Pb[r * 32 + c] = (__bf16)p;
        ls += p;
      }
      m_r = mmax;
      l_r = ls;
      rowv[r] = sc;
    }
    __syncthreads();

    // rescale running O by row correction factors
#pragma unroll
    for (int f = 0; f < 4; ++f)
#pragma unroll
      for (int g = 0; g < 8; ++g) o[f][g] *= rowv[g + hi * 8];

    // P fragment (16x32 A-matrix) from LDS
    U16 pf;
    pf.h[0] = *(const v8bf*)&Pb[l * 32 + hi * 8];
    pf.h[1] = *(const v8bf*)&Pb[l * 32 + hi * 8 + 16];

    // O += P * V  (four 16-col fragments of V)
#pragma unroll
    for (int f = 0; f < 4; ++f) {
      U16 vf;
#pragma unroll
      for (int e = 0; e < 16; ++e)
        vf.e[e] = V[base + (size_t)(kb + hi * 8 + (e & 7) + (e >> 3) * 16) * HD + f * 16 + l];
      o[f] = wmma_bf16(pf.v, vf.v, o[f]);
    }
    __syncthreads();
  }

  if (lane < 16) rowv[lane] = 1.0f / l_r;
  __syncthreads();

#pragma unroll
  for (int f = 0; f < 4; ++f)
#pragma unroll
    for (int g = 0; g < 8; ++g) {
      int m = g + hi * 8;
      int t = q0 + m;
      float val = o[f][g] * rowv[m];
      AO[((size_t)(b * SEQ + t)) * EMB + h * 64 + f * 16 + l] = (__bf16)val;
    }
}

// ---------------------------------------------------------------- launcher
extern "C" void kernel_launch(void* const* d_in, const int* in_sizes, int n_in,
                              void* d_out, int out_size, void* d_ws, size_t ws_size,
                              hipStream_t stream) {
  (void)in_sizes; (void)n_in; (void)out_size; (void)ws_size;
  const float* x    = (const float*)d_in[0];
  const float* wq   = (const float*)d_in[1];
  const float* bq   = (const float*)d_in[2];
  const float* wk   = (const float*)d_in[3];
  const float* bk   = (const float*)d_in[4];
  const float* wv   = (const float*)d_in[5];
  const float* bv   = (const float*)d_in[6];
  const float* wo   = (const float*)d_in[7];
  const float* bo   = (const float*)d_in[8];
  const float* w1   = (const float*)d_in[9];
  const float* b1   = (const float*)d_in[10];
  const float* w2   = (const float*)d_in[11];
  const float* b2   = (const float*)d_in[12];
  const float* ln1g = (const float*)d_in[13];
  const float* ln1b = (const float*)d_in[14];
  const float* ln2g = (const float*)d_in[15];
  const float* ln2b = (const float*)d_in[16];
  const float* aAt  = (const float*)d_in[17];
  const float* aFf  = (const float*)d_in[18];

  char* ws = (char*)d_ws;
  size_t off = 0;
  auto alloc = [&](size_t bytes) -> char* {
    char* p = ws + off;
    off += (bytes + 255) & ~(size_t)255;
    return p;
  };
  __bf16* wqb = (__bf16*)alloc((size_t)EMB * EMB * 2);
  __bf16* wkb = (__bf16*)alloc((size_t)EMB * EMB * 2);
  __bf16* wvb = (__bf16*)alloc((size_t)EMB * EMB * 2);
  __bf16* wob = (__bf16*)alloc((size_t)EMB * EMB * 2);
  __bf16* w1b = (__bf16*)alloc((size_t)EMB * FFDIM * 2);
  __bf16* w2b = (__bf16*)alloc((size_t)FFDIM * EMB * 2);
  __bf16* xn  = (__bf16*)alloc((size_t)MROWS * EMB * 2);
  __bf16* qb  = (__bf16*)alloc((size_t)MROWS * EMB * 2); // [B,H,T,D]
  __bf16* kb  = (__bf16*)alloc((size_t)MROWS * EMB * 2);
  __bf16* vb  = (__bf16*)alloc((size_t)MROWS * EMB * 2);
  __bf16* ao  = (__bf16*)alloc((size_t)MROWS * EMB * 2); // [B,T,E]
  float*  x1  = (float*)alloc((size_t)MROWS * EMB * 4);
  __bf16* x2n = (__bf16*)alloc((size_t)MROWS * EMB * 2);
  __bf16* hb  = (__bf16*)alloc((size_t)MROWS * FFDIM * 2);

  // 1. weights -> bf16
  cvt_bf16_k<<<1024, 256, 0, stream>>>(wq, wqb, EMB * EMB);
  cvt_bf16_k<<<1024, 256, 0, stream>>>(wk, wkb, EMB * EMB);
  cvt_bf16_k<<<1024, 256, 0, stream>>>(wv, wvb, EMB * EMB);
  cvt_bf16_k<<<1024, 256, 0, stream>>>(wo, wob, EMB * EMB);
  cvt_bf16_k<<<2048, 256, 0, stream>>>(w1, w1b, EMB * FFDIM);
  cvt_bf16_k<<<2048, 256, 0, stream>>>(w2, w2b, FFDIM * EMB);

  // 2. LN1
  ln_bf16_k<<<MROWS, 256, 0, stream>>>(x, ln1g, ln1b, xn);

  // 3. QKV projections (q pre-scaled by 1/sqrt(64))
  dim3 gProj(EMB / GN, MROWS / GM);
  gemm_wmma_k<<<gProj, 256, 0, stream>>>(xn, wqb, bq, MROWS, EMB, EMB,
                                         MODE_QKV, 0.125f, aAt, nullptr, nullptr, qb);
  gemm_wmma_k<<<gProj, 256, 0, stream>>>(xn, wkb, bk, MROWS, EMB, EMB,
                                         MODE_QKV, 1.0f, aAt, nullptr, nullptr, kb);
  gemm_wmma_k<<<gProj, 256, 0, stream>>>(xn, wvb, bv, MROWS, EMB, EMB,
                                         MODE_QKV, 1.0f, aAt, nullptr, nullptr, vb);

  // 4. flash attention (causal)
  dim3 gAttn(SEQ / 16, BATCH * NHEAD);
  attn_fwd_k<<<gAttn, 32, 0, stream>>>(qb, kb, vb, ao);

  // 5. output projection + residual: x1 = x + alpha_attn * (ao @ wo + bo)
  gemm_wmma_k<<<gProj, 256, 0, stream>>>(ao, wob, bo, MROWS, EMB, EMB,
                                         MODE_RES, 1.0f, aAt, x, x1, nullptr);

  // 6. LN2
  ln_bf16_k<<<MROWS, 256, 0, stream>>>(x1, ln2g, ln2b, x2n);

  // 7. FF1 + exact GELU
  dim3 gFF1(FFDIM / GN, MROWS / GM);
  gemm_wmma_k<<<gFF1, 256, 0, stream>>>(x2n, w1b, b1, MROWS, FFDIM, EMB,
                                        MODE_GELU, 1.0f, aAt, nullptr, nullptr, hb);

  // 8. FF2 + residual: out = x1 + alpha_ff * (h @ w2 + b2)
  gemm_wmma_k<<<gProj, 256, 0, stream>>>(hb, w2b, b2, MROWS, EMB, FFDIM,
                                         MODE_RES, 1.0f, aFf, x1, (float*)d_out, nullptr);
}
